// PartialVAE_Encoder_62998580297763
// MI455X (gfx1250) — matrix operand order repaired and checked by reading, not compiled, not run on hardware
//
#include <hip/hip_runtime.h>

// ---------------------------------------------------------------------------
// Types
// ---------------------------------------------------------------------------
typedef __attribute__((ext_vector_type(16))) __bf16          v16bf;
typedef __attribute__((ext_vector_type(8)))  float           v8f;
typedef __attribute__((ext_vector_type(4)))  float           f4;
typedef __attribute__((ext_vector_type(4)))  unsigned short  u16x4;

union FragU {
    v16bf          v;
    unsigned short u[16];
};

// float -> bf16 bits, round-to-nearest-even
static __device__ __forceinline__ unsigned short f2bf(float f) {
    union { float f; unsigned int u; } c;
    c.f = f;
    unsigned int u = c.u;
    u += 0x7FFFu + ((u >> 16) & 1u);
    return (unsigned short)(u >> 16);
}

static __device__ __forceinline__ float bf2f(unsigned short u) {
    union { unsigned int i; float f; } c;
    c.i = ((unsigned int)u) << 16;
    return c.f;
}

static __device__ __forceinline__ u16x4 cvt4(f4 a) {
    u16x4 r;
    r.x = f2bf(a.x); r.y = f2bf(a.y); r.z = f2bf(a.z); r.w = f2bf(a.w);
    return r;
}

// ---------------------------------------------------------------------------
// CDNA5 async global->LDS staging (guarded: falls back to reg-staged copies)
// Builtin signature (from hipcc diagnostic): param0 = v4i in AS1 (global),
// param1 = v4i in AS3 (LDS), then imm offset, imm cpol.
// ---------------------------------------------------------------------------
#if __has_builtin(__builtin_amdgcn_global_load_async_to_lds_b128)
#define HAVE_ASYNC 1
typedef __attribute__((__vector_size__(16))) int       v4i;
typedef __attribute__((address_space(1))) v4i*         gas_ptr;
typedef __attribute__((address_space(3))) v4i*         las_ptr;
static __device__ __forceinline__ void async_cp16(const unsigned short* g,
                                                  unsigned short*       l) {
    __builtin_amdgcn_global_load_async_to_lds_b128(
        (gas_ptr)(unsigned long long)g,
        (las_ptr)(unsigned int)(unsigned long long)l, 0, 0);
}
#else
#define HAVE_ASYNC 0
static __device__ __forceinline__ void async_cp16(const unsigned short* g,
                                                  unsigned short*       l) {
    *(u16x4*)l       = *(const u16x4*)g;        // 16B reg-staged copy
    *(u16x4*)(l + 4) = *(const u16x4*)(g + 4);
}
#endif

static __device__ __forceinline__ void wait_async_all() {
#if HAVE_ASYNC
#if __has_builtin(__builtin_amdgcn_s_wait_asynccnt)
    __builtin_amdgcn_s_wait_asynccnt(0);
#else
    asm volatile("s_wait_asynccnt 0x0" ::: "memory");
#endif
#endif
}

// ---------------------------------------------------------------------------
// fp32 -> bf16 bulk convert (prepass over weights / emb table)
// ---------------------------------------------------------------------------
__global__ __launch_bounds__(256) void convert_f32_bf16(
    const float* __restrict__ src, unsigned short* __restrict__ dst, int n4)
{
    int i = blockIdx.x * blockDim.x + threadIdx.x;
    if (i < n4) ((u16x4*)dst)[i] = cvt4(((const f4*)src)[i]);
}

// ---------------------------------------------------------------------------
// GEMM: C[M,N] = act(A[M,Kd] @ W[N,Kd]^T + bias[N]); A, W bf16 in memory,
// fp32 accumulate via v_wmma_f32_16x16x32_bf16. Block tile 128x64, 256
// threads (8 waves); each wave: 32x32 output = 4 accumulators, 4 WMMA per
// K-step. Double-buffered LDS, async global->LDS staging when available.
// Output: bf16 (Cb != null) or fp32 (Cf). Requires M%128==0, N%64==0,
// Kd%32==0 (true for all uses here).
// ---------------------------------------------------------------------------
#define LSTR 40  // padded LDS row stride in ushorts (80 B)

__global__ __launch_bounds__(256) void gemm_bf16_wmma(
    const unsigned short* __restrict__ A, const unsigned short* __restrict__ W,
    const float* __restrict__ bias, float* __restrict__ Cf,
    unsigned short* __restrict__ Cb, int M, int N, int Kd, int do_relu)
{
    __shared__ unsigned short As[2][128 * LSTR];
    __shared__ unsigned short Bs[2][64 * LSTR];

    const int tid  = threadIdx.x;
    const int wave = tid >> 5;
    const int lane = tid & 31;
    const int hi   = lane >> 4;   // half-wave 0/1
    const int lr   = lane & 15;

    const int bm = blockIdx.y * 128;
    const int bn = blockIdx.x * 64;
    const int wm = (wave & 3) * 32;   // wave M offset in tile
    const int wn = (wave >> 2) * 32;  // wave N offset in tile

    // loader: 16B chunks; A tile 128x32 = 512 chunks (2/thread), B 64x32 = 256
    const int lrow = tid >> 2;
    const int lcol = (tid & 3) * 8;

    auto stage = [&](int k0, int s) {
        async_cp16(A + (size_t)(bm + lrow) * Kd + k0 + lcol,
                   &As[s][lrow * LSTR + lcol]);
        async_cp16(A + (size_t)(bm + lrow + 64) * Kd + k0 + lcol,
                   &As[s][(lrow + 64) * LSTR + lcol]);
        async_cp16(W + (size_t)(bn + lrow) * Kd + k0 + lcol,
                   &Bs[s][lrow * LSTR + lcol]);
    };

    v8f acc[2][2];
    #pragma unroll
    for (int mi = 0; mi < 2; ++mi)
        #pragma unroll
        for (int ni = 0; ni < 2; ++ni)
            acc[mi][ni] = (v8f){0.f, 0.f, 0.f, 0.f, 0.f, 0.f, 0.f, 0.f};

    stage(0, 0);
    wait_async_all();
    __syncthreads();

    int cur = 0;
    for (int k0 = 0; k0 < Kd; k0 += 32) {
        const int nxt = cur ^ 1;
        if (k0 + 32 < Kd) stage(k0 + 32, nxt);

        // ---- fragments per CDNA5 wave32 WMMA layout ----
        // A (16x32): half 0 -> K {0..7,16..23}, half 1 -> K {8..15,24..31}
        FragU fa[2], fb[2];
        #pragma unroll
        for (int mi = 0; mi < 2; ++mi) {
            const unsigned short* ar = &As[cur][(wm + mi * 16 + lr) * LSTR];
            *(u16x4*)&fa[mi].u[0]  = *(const u16x4*)&ar[8 * hi];
            *(u16x4*)&fa[mi].u[4]  = *(const u16x4*)&ar[8 * hi + 4];
            *(u16x4*)&fa[mi].u[8]  = *(const u16x4*)&ar[16 + 8 * hi];
            *(u16x4*)&fa[mi].u[12] = *(const u16x4*)&ar[16 + 8 * hi + 4];
        }
        // B (32x16): lane = column, half 0 -> K {0..7,16..23}, half 1 -> K {8..15,24..31}
        #pragma unroll
        for (int ni = 0; ni < 2; ++ni) {
            const unsigned short* br = &Bs[cur][(wn + ni * 16 + lr) * LSTR + 8 * hi];
            #pragma unroll
            for (int q = 0; q < 4; ++q)
                *(u16x4*)&fb[ni].u[4 * q] = *(const u16x4*)&br[4 * q + 8 * (q >> 1)];
        }
        #pragma unroll
        for (int mi = 0; mi < 2; ++mi)
            #pragma unroll
            for (int ni = 0; ni < 2; ++ni)
                acc[mi][ni] = __builtin_amdgcn_wmma_f32_16x16x32_bf16(
                    false, fa[mi].v, false, fb[ni].v, (short)0, acc[mi][ni],
                    false, false);

        wait_async_all();
        __syncthreads();
        cur = nxt;
    }

    // ---- epilogue: C layout row = vgpr + 8*hi, col = lane%16 ----
    const int   col0 = bn + wn + lr;
    const int   col1 = col0 + 16;
    const float bb0  = bias[col0];
    const float bb1  = bias[col1];
    #pragma unroll
    for (int mi = 0; mi < 2; ++mi) {
        const int row0 = bm + wm + mi * 16 + 8 * hi;
        #pragma unroll
        for (int r = 0; r < 8; ++r) {
            float v0 = acc[mi][0][r] + bb0;
            float v1 = acc[mi][1][r] + bb1;
            if (do_relu) { v0 = fmaxf(v0, 0.f); v1 = fmaxf(v1, 0.f); }
            const size_t o0 = (size_t)(row0 + r) * N + col0;
            const size_t o1 = (size_t)(row0 + r) * N + col1;
            if (Cb) { Cb[o0] = f2bf(v0); Cb[o1] = f2bf(v1); }
            else    { Cf[o0] = v0;       Cf[o1] = v1; }
        }
    }
}

// ---------------------------------------------------------------------------
// pooled[b,k] = sum_d mask[b,d] * T[x[b,d], k]  (T bf16, fp32 accumulate,
// bf16 result). One block per b; 128 lanes x 4 cols = 512. Index/mask reads
// are block-uniform (scalar path); masked rows skip the load entirely.
// One prefetch per 128B line, 8 rows ahead, to cover L2 latency.
// ---------------------------------------------------------------------------
__global__ __launch_bounds__(128) void gather_pool(
    const unsigned short* __restrict__ T, const int* __restrict__ x,
    const unsigned char* __restrict__ mask, unsigned short* __restrict__ pooled)
{
    const int b = blockIdx.x;
    const int t = threadIdx.x;
    const int* __restrict__           xr = x + b * 1024;
    const unsigned char* __restrict__ mr = mask + b * 1024;

    f4 acc = {0.f, 0.f, 0.f, 0.f};
    #pragma unroll 4
    for (int d = 0; d < 1024; ++d) {
        if (d + 8 < 1024 && ((t & 15) == 0) && mr[d + 8])
            __builtin_prefetch(T + (size_t)xr[d + 8] * 512 + 4 * t, 0, 0);
        if (mr[d]) {
            u16x4 v = *(const u16x4*)(T + (size_t)xr[d] * 512 + 4 * t);
            acc.x += bf2f(v.x); acc.y += bf2f(v.y);
            acc.z += bf2f(v.z); acc.w += bf2f(v.w);
        }
    }
    *(u16x4*)(pooled + b * 512 + 4 * t) = cvt4(acc);
}

// ---------------------------------------------------------------------------
// z = mu + eps * exp(0.5 * logvar)
// ---------------------------------------------------------------------------
__global__ __launch_bounds__(256) void reparam_kernel(
    const float* __restrict__ mu, const float* __restrict__ lv,
    const float* __restrict__ eps, float* __restrict__ z, int n)
{
    int i = blockIdx.x * blockDim.x + threadIdx.x;
    if (i < n) z[i] = mu[i] + eps[i] * __expf(0.5f * lv[i]);
}

// ---------------------------------------------------------------------------
// Launch
// ---------------------------------------------------------------------------
extern "C" void kernel_launch(void* const* d_in, const int* in_sizes, int n_in,
                              void* d_out, int out_size, void* d_ws, size_t ws_size,
                              hipStream_t stream) {
    const int*           x    = (const int*)d_in[0];            // [256,1024]
    const unsigned char* mask = (const unsigned char*)d_in[1];  // [256,1024] bool
    const float*         eps  = (const float*)d_in[2];          // [256,64]
    const float*         emb  = (const float*)d_in[3];          // [2048,128]
    const float*         Wp   = (const float*)d_in[4];          // [512,128]
    const float*         bp   = (const float*)d_in[5];
    const float*         W1   = (const float*)d_in[6];          // [1024,512]
    const float*         b1   = (const float*)d_in[7];
    const float*         W2   = (const float*)d_in[8];          // [512,1024]
    const float*         b2   = (const float*)d_in[9];
    const float*         Wmu  = (const float*)d_in[10];         // [64,512]
    const float*         bmu  = (const float*)d_in[11];
    const float*         Wlv  = (const float*)d_in[12];         // [64,512]
    const float*         blv  = (const float*)d_in[13];

    // workspace carve-out (all bf16 / ushort)
    unsigned short* Tb   = (unsigned short*)d_ws;    // [2048,512]
    unsigned short* embB = Tb + 2048 * 512;          // [2048,128]
    unsigned short* WpB  = embB + 2048 * 128;        // [512,128]
    unsigned short* W1B  = WpB + 512 * 128;          // [1024,512]
    unsigned short* W2B  = W1B + 1024 * 512;         // [512,1024]
    unsigned short* WmB  = W2B + 512 * 1024;         // [64,512]
    unsigned short* WlB  = WmB + 64 * 512;           // [64,512]
    unsigned short* plB  = WlB + 64 * 512;           // pooled [256,512]
    unsigned short* h1B  = plB + 256 * 512;          // [256,1024]
    unsigned short* h2B  = h1B + 256 * 1024;         // [256,512]

    float* out = (float*)d_out;                      // concat(z, mu, logvar)
    float* z   = out;
    float* mu  = out + 256 * 64;
    float* lv  = out + 2 * 256 * 64;

    dim3 blk(256);

    // prepass: fp32 -> bf16 for all matrices feeding WMMA
    auto cvt = [&](const float* s, unsigned short* d, int n) {
        convert_f32_bf16<<<(n / 4 + 255) / 256, blk, 0, stream>>>(s, d, n / 4);
    };
    cvt(emb, embB, 2048 * 128);
    cvt(Wp,  WpB,  512 * 128);
    cvt(W1,  W1B,  1024 * 512);
    cvt(W2,  W2B,  512 * 1024);
    cvt(Wmu, WmB,  64 * 512);
    cvt(Wlv, WlB,  64 * 512);

    // T = relu(emb_table @ W_pnnn^T + b_pnnn) -> bf16 : M=2048,N=512,K=128
    gemm_bf16_wmma<<<dim3(512 / 64, 2048 / 128), blk, 0, stream>>>(
        embB, WpB, bp, nullptr, Tb, 2048, 512, 128, 1);

    // pooled[b,:] = sum_d mask * T[x[b,d],:]  -> bf16
    gather_pool<<<256, 128, 0, stream>>>(Tb, x, mask, plB);

    // h1 = relu(pooled @ W_fc1^T + b_fc1) -> bf16 : M=256,N=1024,K=512
    gemm_bf16_wmma<<<dim3(1024 / 64, 256 / 128), blk, 0, stream>>>(
        plB, W1B, b1, nullptr, h1B, 256, 1024, 512, 1);

    // h2 = relu(h1 @ W_fc2^T + b_fc2) -> bf16 : M=256,N=512,K=1024
    gemm_bf16_wmma<<<dim3(512 / 64, 256 / 128), blk, 0, stream>>>(
        h1B, W2B, b2, nullptr, h2B, 256, 512, 1024, 1);

    // mu / logvar -> fp32 output slices : M=256,N=64,K=512
    gemm_bf16_wmma<<<dim3(1, 2), blk, 0, stream>>>(
        h2B, WmB, bmu, mu, nullptr, 256, 64, 512, 0);
    gemm_bf16_wmma<<<dim3(1, 2), blk, 0, stream>>>(
        h2B, WlB, blv, lv, nullptr, 256, 64, 512, 0);

    // z = mu + eps * exp(0.5 * logvar)
    reparam_kernel<<<(256 * 64 + 255) / 256, blk, 0, stream>>>(mu, lv, eps, z, 256 * 64);
}
